// AttentionWithEinops_11553462026636
// MI455X (gfx1250) — matrix operand order repaired and checked by reading, not compiled, not run on hardware
//
#include <hip/hip_runtime.h>
#include <hip/hip_bf16.h>

// ---------------------------------------------------------------------------
// AttentionWithEinops for MI455X (gfx1250): bf16 WMMA GEMMs + flash attention.
// B=2, S=2048, D=1024, H=16, Dh=64.
// Pipeline: convert(fp32->bf16) -> QKV proj (WMMA) -> flash attn (WMMA) ->
//           out proj (WMMA). All matrix math on v_wmma_f32_16x16x32_bf16.
// ---------------------------------------------------------------------------

typedef __bf16 bf16_t;
typedef __attribute__((ext_vector_type(16))) __bf16 v16bf;
typedef __attribute__((ext_vector_type(8)))  __bf16 v8bf;
typedef __attribute__((ext_vector_type(8)))  float  v8f;

#define D_MODEL 1024
#define SEQ     2048
#define NHEAD   16
#define DHEAD   64
#define BATCH   2
#define NTOK    (BATCH * SEQ)        // 4096
#define LDSP    72                    // padded LDS row stride (bf16 elems)

static __device__ inline v8f wmma_bf16(v16bf a, v16bf b, v8f c) {
  // D = A(16x32) * B(32x16) + C, fp32 accumulate
  return __builtin_amdgcn_wmma_f32_16x16x32_bf16(
      /*neg_a=*/false, a, /*neg_b=*/false, b,
      /*c_mod=*/(short)0, c, /*reuse_a=*/false, /*reuse_b=*/false);
}

// A fragment (16x32, M x K) from bf16 row-major [*, ld].
// ISA layout: lane<16 -> row=lane, halves 0..7 = K+0..7, halves 8..15 = K+16..23
//             lane>=16 -> same row group, K offset +8.
static __device__ inline v16bf load_a_bf16(const bf16_t* __restrict__ src, int ld,
                                           int m0, int k0, int lane) {
  const bf16_t* p = src + (long)(m0 + (lane & 15)) * ld + k0 + ((lane >> 4) << 3);
  v8bf lo = *(const v8bf*)p;          // 16B
  v8bf hi = *(const v8bf*)(p + 16);   // 16B
  v16bf a;
#pragma unroll
  for (int i = 0; i < 8; ++i) { a[i] = lo[i]; a[i + 8] = hi[i]; }
  return a;
}

// B fragment (32x16, K x N) where B[k][n] = src[n][k] (src row-major over k).
// ISA layout: lane holds col N=lane%16, halves 0..7 = K+0..7, 8..15 = K+16..23 (+8 upper half).
static __device__ inline v16bf load_bt_bf16(const bf16_t* __restrict__ src, int ld,
                                            int k0, int n0, int lane) {
  const bf16_t* p = src + (long)(n0 + (lane & 15)) * ld + k0 + ((lane >> 4) << 3);
  v8bf lo = *(const v8bf*)p;
  v8bf hi = *(const v8bf*)(p + 16);
  v16bf b;
#pragma unroll
  for (int i = 0; i < 8; ++i) { b[i] = lo[i]; b[i + 8] = hi[i]; }
  return b;
}

// ---------------------------------------------------------------------------
// Kernel 0: vectorized fp32 -> bf16 conversion (8 elems / thread).
// ---------------------------------------------------------------------------
__global__ __launch_bounds__(256) void convert_bf16_kernel(
    const float* __restrict__ src, bf16_t* __restrict__ dst) {
  const long idx = ((long)blockIdx.x * 256 + threadIdx.x) * 8;
  v8f v = *(const v8f*)(src + idx);
  v8bf o;
#pragma unroll
  for (int i = 0; i < 8; ++i) o[i] = (bf16_t)v[i];
  *(v8bf*)(dst + idx) = o;
}

// ---------------------------------------------------------------------------
// Kernel 1: fused Q/K/V projection. grid=(M/64, N/256, 3), block=256 (8 waves).
// q[m,he] = sum_d X[m,d] * W[he,d]   (W row-major over d == Bt layout)
// z=0 -> Q [b,h,s,e]; z=1 -> K [b,h,s,e]; z=2 -> V transposed [b,h,e,s]
// ---------------------------------------------------------------------------
__global__ __launch_bounds__(256) void qkv_proj_kernel(
    const bf16_t* __restrict__ X, const bf16_t* __restrict__ Wq,
    const bf16_t* __restrict__ Wk, const bf16_t* __restrict__ Wv,
    bf16_t* __restrict__ qbuf, bf16_t* __restrict__ kbuf,
    bf16_t* __restrict__ vtbuf) {
  const int lane = threadIdx.x & 31;
  const int w    = threadIdx.x >> 5;
  const int wm   = w >> 2, wn = w & 3;
  const int m0 = blockIdx.x * 64 + wm * 32;
  const int n0 = blockIdx.y * 256 + wn * 64;
  const bf16_t* W = (blockIdx.z == 0) ? Wq : (blockIdx.z == 1) ? Wk : Wv;

  v8f acc[2][4];
#pragma unroll
  for (int i = 0; i < 2; ++i)
#pragma unroll
    for (int t = 0; t < 4; ++t) acc[i][t] = (v8f){0,0,0,0,0,0,0,0};

  for (int k0 = 0; k0 < D_MODEL; k0 += 32) {
    v16bf a0 = load_a_bf16(X, D_MODEL, m0,      k0, lane);
    v16bf a1 = load_a_bf16(X, D_MODEL, m0 + 16, k0, lane);
#pragma unroll
    for (int t = 0; t < 4; ++t) {
      v16bf b = load_bt_bf16(W, D_MODEL, k0, n0 + 16 * t, lane);
      acc[0][t] = wmma_bf16(a0, b, acc[0][t]);
      acc[1][t] = wmma_bf16(a1, b, acc[1][t]);
    }
  }

  bf16_t* dst = (blockIdx.z == 0) ? qbuf : (blockIdx.z == 1) ? kbuf : vtbuf;
  const bool vtrans = (blockIdx.z == 2);
  const int halfoff = (lane >> 4) << 3;
#pragma unroll
  for (int i = 0; i < 2; ++i) {
#pragma unroll
    for (int t = 0; t < 4; ++t) {
      const int col = n0 + 16 * t + (lane & 15);
      const int h = col >> 6, e = col & 63;
#pragma unroll
      for (int r = 0; r < 8; ++r) {
        const int m = m0 + 16 * i + halfoff + r;
        const int b = m >> 11, s = m & (SEQ - 1);
        const long idx = vtrans
            ? ((long)((b * NHEAD + h) * DHEAD + e) * SEQ + s)
            : ((long)((b * NHEAD + h) * SEQ + s) * DHEAD + e);
        dst[idx] = (bf16_t)acc[i][t][r];
      }
    }
  }
}

// ---------------------------------------------------------------------------
// Kernel 2: flash attention with causal mask.
// grid=(S/64, B*H), block=128 (4 waves); wave owns 16 query rows.
// ---------------------------------------------------------------------------
__global__ __launch_bounds__(128) void attn_kernel(
    const bf16_t* __restrict__ qbuf, const bf16_t* __restrict__ kbuf,
    const bf16_t* __restrict__ vtbuf, bf16_t* __restrict__ attn) {
  __shared__ bf16_t lds[4 * 16 * LDSP];   // per-wave 16x64 P tile, padded
  const int lane = threadIdx.x & 31;
  const int w    = threadIdx.x >> 5;
  const int bh   = blockIdx.y;
  const int bidx = bh >> 4, h = bh & 15;
  const int q0   = blockIdx.x * 64 + w * 16;   // query row base (s index)
  const int halfoff = (lane >> 4) << 3;

  const bf16_t* Q  = qbuf  + (long)bh * SEQ * DHEAD;
  const bf16_t* K  = kbuf  + (long)bh * SEQ * DHEAD;
  const bf16_t* Vt = vtbuf + (long)bh * DHEAD * SEQ;
  bf16_t* plds = lds + w * 16 * LDSP;

  const float scale = 0.125f;   // 1/sqrt(64)

  // Q fragments are loop-invariant (Dh=64 -> two K=32 fragments)
  v16bf aq0 = load_a_bf16(Q, DHEAD, q0, 0,  lane);
  v16bf aq1 = load_a_bf16(Q, DHEAD, q0, 32, lane);

  v8f acc_o[4];
#pragma unroll
  for (int t = 0; t < 4; ++t) acc_o[t] = (v8f){0,0,0,0,0,0,0,0};
  float m_i[8], l_i[8];
#pragma unroll
  for (int r = 0; r < 8; ++r) { m_i[r] = -3.0e38f; l_i[r] = 0.0f; }

  const int ntiles = blockIdx.x + 1;   // causal: only key tiles up to the diagonal
  for (int jt = 0; jt < ntiles; ++jt) {
    const int j0 = jt * 64;

    // ---- S = Q * K^T (16x64 tile per wave) ----
    v8f s_acc[4];
#pragma unroll
    for (int t = 0; t < 4; ++t) {
      s_acc[t] = (v8f){0,0,0,0,0,0,0,0};
      v16bf bk0 = load_bt_bf16(K, DHEAD, 0,  j0 + 16 * t, lane);
      s_acc[t] = wmma_bf16(aq0, bk0, s_acc[t]);
      v16bf bk1 = load_bt_bf16(K, DHEAD, 32, j0 + 16 * t, lane);
      s_acc[t] = wmma_bf16(aq1, bk1, s_acc[t]);
    }

    // ---- scale + causal mask + online softmax (per row r) ----
#pragma unroll
    for (int r = 0; r < 8; ++r) {
      const int row = q0 + halfoff + r;
      float mx = m_i[r];
#pragma unroll
      for (int t = 0; t < 4; ++t) {
        const int col = j0 + 16 * t + (lane & 15);
        float v = s_acc[t][r] * scale;
        v = (col <= row) ? v : -3.0e38f;
        s_acc[t][r] = v;
        mx = fmaxf(mx, v);
      }
      // row reduction across the 16 lanes of this half-wave
#pragma unroll
      for (int off = 8; off >= 1; off >>= 1) mx = fmaxf(mx, __shfl_xor(mx, off, 32));
      const float alpha = __expf(m_i[r] - mx);
      float lsum = 0.0f;
#pragma unroll
      for (int t = 0; t < 4; ++t) {
        const float p = __expf(s_acc[t][r] - mx);
        s_acc[t][r] = p;
        lsum += p;
      }
#pragma unroll
      for (int off = 8; off >= 1; off >>= 1) lsum += __shfl_xor(lsum, off, 32);
      m_i[r] = mx;
      l_i[r] = l_i[r] * alpha + lsum;
#pragma unroll
      for (int t = 0; t < 4; ++t) acc_o[t][r] *= alpha;
    }

    // ---- C-layout P -> LDS (row major) -> A-layout fragments ----
#pragma unroll
    for (int t = 0; t < 4; ++t)
#pragma unroll
      for (int r = 0; r < 8; ++r)
        plds[(halfoff + r) * LDSP + 16 * t + (lane & 15)] = (bf16_t)s_acc[t][r];

    v16bf ap0 = load_a_bf16(plds, LDSP, 0, 0,  lane);
    v16bf ap1 = load_a_bf16(plds, LDSP, 0, 32, lane);

    // ---- O += P * V  (V stored transposed [e][s] -> contiguous-K B frags) ----
#pragma unroll
    for (int t = 0; t < 4; ++t) {
      v16bf bv0 = load_bt_bf16(Vt, SEQ, j0,      16 * t, lane);
      acc_o[t] = wmma_bf16(ap0, bv0, acc_o[t]);
      v16bf bv1 = load_bt_bf16(Vt, SEQ, j0 + 32, 16 * t, lane);
      acc_o[t] = wmma_bf16(ap1, bv1, acc_o[t]);
    }
  }

  // ---- normalize and write attn_out [b, s, h*64+e] as bf16 ----
#pragma unroll
  for (int t = 0; t < 4; ++t) {
#pragma unroll
    for (int r = 0; r < 8; ++r) {
      const int s = q0 + halfoff + r;
      const int e = 16 * t + (lane & 15);
      const float o = acc_o[t][r] / l_i[r];
      attn[(long)(bidx * SEQ + s) * (NHEAD * DHEAD) + h * DHEAD + e] = (bf16_t)o;
    }
  }
}

// ---------------------------------------------------------------------------
// Kernel 3: W_O [he, d] fp32 -> wot [d, he] bf16 (so out-proj B frags are
// contiguous along K).
// ---------------------------------------------------------------------------
__global__ __launch_bounds__(256) void wo_transpose_kernel(
    const float* __restrict__ Wo, bf16_t* __restrict__ wot) {
  const int idx = blockIdx.x * 256 + threadIdx.x;   // 1M elements
  const int k = idx >> 10;    // he
  const int n = idx & 1023;   // d
  wot[(long)n * D_MODEL + k] = (bf16_t)Wo[idx];
}

// ---------------------------------------------------------------------------
// Kernel 4: output projection. out[m,d] = sum_he attn[m,he] * Wo[he,d].
// grid=(M/64, N/256), block=256 (8 waves), fp32 output.
// ---------------------------------------------------------------------------
__global__ __launch_bounds__(256) void out_proj_kernel(
    const bf16_t* __restrict__ attn, const bf16_t* __restrict__ wot,
    float* __restrict__ out) {
  const int lane = threadIdx.x & 31;
  const int w    = threadIdx.x >> 5;
  const int wm   = w >> 2, wn = w & 3;
  const int m0 = blockIdx.x * 64 + wm * 32;
  const int n0 = blockIdx.y * 256 + wn * 64;

  v8f acc[2][4];
#pragma unroll
  for (int i = 0; i < 2; ++i)
#pragma unroll
    for (int t = 0; t < 4; ++t) acc[i][t] = (v8f){0,0,0,0,0,0,0,0};

  for (int k0 = 0; k0 < D_MODEL; k0 += 32) {
    v16bf a0 = load_a_bf16(attn, D_MODEL, m0,      k0, lane);
    v16bf a1 = load_a_bf16(attn, D_MODEL, m0 + 16, k0, lane);
#pragma unroll
    for (int t = 0; t < 4; ++t) {
      v16bf b = load_bt_bf16(wot, D_MODEL, k0, n0 + 16 * t, lane);
      acc[0][t] = wmma_bf16(a0, b, acc[0][t]);
      acc[1][t] = wmma_bf16(a1, b, acc[1][t]);
    }
  }

  const int halfoff = (lane >> 4) << 3;
#pragma unroll
  for (int i = 0; i < 2; ++i) {
#pragma unroll
    for (int t = 0; t < 4; ++t) {
      const int col = n0 + 16 * t + (lane & 15);
#pragma unroll
      for (int r = 0; r < 8; ++r) {
        const int m = m0 + 16 * i + halfoff + r;
        out[(long)m * D_MODEL + col] = acc[i][t][r];
      }
    }
  }
}

// ---------------------------------------------------------------------------
extern "C" void kernel_launch(void* const* d_in, const int* in_sizes, int n_in,
                              void* d_out, int out_size, void* d_ws, size_t ws_size,
                              hipStream_t stream) {
  const float* residual = (const float*)d_in[0];
  const float* Wq = (const float*)d_in[1];
  const float* Wk = (const float*)d_in[2];
  const float* Wv = (const float*)d_in[3];
  const float* Wo = (const float*)d_in[4];
  float* out = (float*)d_out;

  char* ws = (char*)d_ws;
  const size_t MB = 1024 * 1024;
  bf16_t* qbuf  = (bf16_t*)(ws);              //  8 MB: [b,h,s,e] bf16
  bf16_t* kbuf  = (bf16_t*)(ws + 8  * MB);    //  8 MB: [b,h,s,e] bf16
  bf16_t* vtbuf = (bf16_t*)(ws + 16 * MB);    //  8 MB: [b,h,e,s] bf16
  bf16_t* attnb = (bf16_t*)(ws + 24 * MB);    //  8 MB: [b*s, h*e] bf16
  bf16_t* wot   = (bf16_t*)(ws + 32 * MB);    //  2 MB: [d, he] bf16
  bf16_t* xb    = (bf16_t*)(ws + 34 * MB);    //  8 MB: residual bf16
  bf16_t* wqb   = (bf16_t*)(ws + 42 * MB);    //  2 MB
  bf16_t* wkb   = (bf16_t*)(ws + 44 * MB);    //  2 MB
  bf16_t* wvb   = (bf16_t*)(ws + 46 * MB);    //  2 MB

  // fp32 -> bf16 pre-conversion (keeps GEMM hot loops conversion-free)
  convert_bf16_kernel<<<(NTOK * D_MODEL) / 2048, 256, 0, stream>>>(residual, xb);
  convert_bf16_kernel<<<(D_MODEL * D_MODEL) / 2048, 256, 0, stream>>>(Wq, wqb);
  convert_bf16_kernel<<<(D_MODEL * D_MODEL) / 2048, 256, 0, stream>>>(Wk, wkb);
  convert_bf16_kernel<<<(D_MODEL * D_MODEL) / 2048, 256, 0, stream>>>(Wv, wvb);
  wo_transpose_kernel<<<(D_MODEL * D_MODEL) / 256, 256, 0, stream>>>(Wo, wot);

  dim3 gproj(NTOK / 64, D_MODEL / 256, 3);
  qkv_proj_kernel<<<gproj, 256, 0, stream>>>(xb, wqb, wkb, wvb,
                                             qbuf, kbuf, vtbuf);

  dim3 gattn(SEQ / 64, BATCH * NHEAD);
  attn_kernel<<<gattn, 128, 0, stream>>>(qbuf, kbuf, vtbuf, attnb);

  dim3 gout(NTOK / 64, D_MODEL / 256);
  out_proj_kernel<<<gout, 256, 0, stream>>>(attnb, wot, out);
}
